// FlaxDbrxExpertGLU_29472065585699
// MI455X (gfx1250) — matrix operand order repaired and checked by reading, compiled, not run on hardware
//
#include <hip/hip_runtime.h>
#include <hip/hip_bf16.h>

// CDNA5 / gfx1250 fused MoE expert FFN, round 2:
//  pass 0: one-shot fp32->bf16 conversion of x, w1[e], v1[e]; w2[e] converted
//          AND transposed to d-major so all GEMM staging is k-contiguous bf16.
//  pass 1: h = silu(x @ w1^T) * (x @ v1^T), bf16 WMMA, double-buffered LDS
//          tiles filled with GLOBAL_LOAD_ASYNC_TO_LDS_B128 (ASYNCcnt pipelined).
//  pass 2: out = h @ w2 (fp32 out), same async double-buffered scheme.

typedef __attribute__((ext_vector_type(16))) __bf16 v16bf;
typedef __attribute__((ext_vector_type(8)))  float  v8f;

constexpr int FFN  = 8192;
constexpr int DMOD = 2048;
constexpr int TSEQ = 4096;

constexpr int BM  = 128;
constexpr int BN  = 128;
constexpr int BK  = 64;
constexpr int LDK = BK + 8;   // 72 bf16 = 144 B row stride: 16B-aligned, bank padding

union Frag { uint4 u[2]; v16bf v; };

// A fragment (16x32 bf16): lane m = lane&15; lanes 0-15 hold K {ks..+7, ks+16..+23},
// lanes 16-31 hold K {ks+8..+15, ks+24..+31}  (CDNA5 ISA 16-bit A layout).
__device__ __forceinline__ v16bf load_a_frag(const __bf16* base, int lane, int ks) {
    const int lm = lane & 15, lh = lane >> 4;
    const __bf16* p = base + lm * LDK + ks + lh * 8;
    Frag f;
    f.u[0] = *(const uint4*)(p);
    f.u[1] = *(const uint4*)(p + 16);
    return f.v;
}

// B fragment (32x16 bf16): lane n = lane&15; lanes 0-15 hold K {ks..+7, ks+16..+23},
// lanes 16-31 hold K {ks+8..+15, ks+24..+31} (same K order as A).
__device__ __forceinline__ v16bf load_b_frag(const __bf16* base, int lane, int ks) {
    const int ln = lane & 15, lh = lane >> 4;
    const __bf16* p = base + ln * LDK + ks + lh * 8;
    Frag f;
    f.u[0] = *(const uint4*)(p);
    f.u[1] = *(const uint4*)(p + 16);
    return f.v;
}

__device__ __forceinline__ v8f wmma_bf16(v16bf a, v16bf b, v8f c) {
    return __builtin_amdgcn_wmma_f32_16x16x32_bf16(false, a, false, b, (short)0, c,
                                                   false, false);
}

// Async VMEM -> LDS copy of 16 bytes per lane (ASYNCcnt-tracked, no VGPR data).
// LDS byte offset = low 32 bits of the generic pointer (LDS aperture: addr[31:0]).
__device__ __forceinline__ void async_copy16(void* lds, const void* gaddr) {
    asm volatile("global_load_async_to_lds_b128 %0, %1, off"
                 :: "v"((unsigned)(unsigned long long)lds),
                    "v"((unsigned long long)gaddr)
                 : "memory");
}

// ---------------------------------------------------------------------------
// Pass 0a: elementwise fp32 -> bf16 (optionally offset by expert slab)
// ---------------------------------------------------------------------------
__global__ __launch_bounds__(256)
void convert_bf16_kernel(const float* __restrict__ src, __bf16* __restrict__ dst,
                         int n4, const int* __restrict__ eidx, size_t eoffStride) {
    const size_t i = (size_t)blockIdx.x * blockDim.x + threadIdx.x;
    if (i >= (size_t)n4) return;
    const size_t eoff = (size_t)eidx[0] * eoffStride;
    const float4 f = *(const float4*)(src + eoff + i * 4);
    union { __bf16 b[4]; uint2 u; } o;
    o.b[0] = (__bf16)f.x; o.b[1] = (__bf16)f.y; o.b[2] = (__bf16)f.z; o.b[3] = (__bf16)f.w;
    *(uint2*)(dst + i * 4) = o.u;
}

// ---------------------------------------------------------------------------
// Pass 0b: w2[e] (FFN x D, f-major) -> bf16 transposed w2t (D x FFN, d-major)
// ---------------------------------------------------------------------------
__global__ __launch_bounds__(256)
void transpose_convert_kernel(const float* __restrict__ w2, __bf16* __restrict__ w2t,
                              const int* __restrict__ eidx) {
    __shared__ float tile[32][33];
    const size_t eoff = (size_t)eidx[0] * (size_t)FFN * (size_t)DMOD;
    const int f0 = blockIdx.x * 32;             // along FFN
    const int d0 = blockIdx.y * 32;             // along D
    const int tx = threadIdx.x & 31, ty = threadIdx.x >> 5;   // 32 x 8
    #pragma unroll
    for (int i = 0; i < 4; ++i)
        tile[ty + 8 * i][tx] = w2[eoff + (size_t)(f0 + ty + 8 * i) * DMOD + d0 + tx];
    __syncthreads();
    #pragma unroll
    for (int i = 0; i < 4; ++i)
        w2t[(size_t)(d0 + ty + 8 * i) * FFN + f0 + tx] = (__bf16)tile[tx][ty + 8 * i];
}

// ---------------------------------------------------------------------------
// Pass 1: gate/up projections + SiLU*up; h stored bf16 row-major (T x FFN)
// ---------------------------------------------------------------------------
__global__ __launch_bounds__(256)
void gateup_kernel(const __bf16* __restrict__ xb, const __bf16* __restrict__ w1b,
                   const __bf16* __restrict__ v1b, __bf16* __restrict__ hbuf) {
    __shared__ alignas(16) __bf16 sA[2][BM * LDK];
    __shared__ alignas(16) __bf16 sW[2][BN * LDK];
    __shared__ alignas(16) __bf16 sV[2][BN * LDK];

    const int tid   = threadIdx.x;
    const int lane  = tid & 31;
    const int wave  = tid >> 5;
    const int waveM = wave >> 2;      // 0..1  -> 64-row patch
    const int waveN = wave & 3;       // 0..3  -> 32-col patch
    const int mBlock = blockIdx.y * BM;
    const int nBlock = blockIdx.x * BN;

    v8f accG[4][2] = {};
    v8f accU[4][2] = {};

    // 12 async b128 copies per thread per K-step (3 tiles x 16KB / 256 lanes)
    auto issue = [&](int buf, int kb) {
        #pragma unroll
        for (int p = 0; p < 4; ++p) {
            const int row = (tid >> 3) + p * 32;   // 0..127
            const int ch  = (tid & 7) * 8;         // 0..56 bf16 (16B granules)
            async_copy16(&sA[buf][row * LDK + ch],
                         xb  + (size_t)(mBlock + row) * DMOD + kb + ch);
            async_copy16(&sW[buf][row * LDK + ch],
                         w1b + (size_t)(nBlock + row) * DMOD + kb + ch);
            async_copy16(&sV[buf][row * LDK + ch],
                         v1b + (size_t)(nBlock + row) * DMOD + kb + ch);
        }
    };

    issue(0, 0);
    const int nK = DMOD / BK;         // 32
    for (int ki = 0; ki < nK; ++ki) {
        const int cur = ki & 1;
        if (ki + 1 < nK) {
            issue(cur ^ 1, (ki + 1) * BK);
            asm volatile("s_wait_asynccnt 0xc" ::: "memory");  // next tile may stay in flight
        } else {
            asm volatile("s_wait_asynccnt 0x0" ::: "memory");
        }
        __syncthreads();

        #pragma unroll
        for (int ks = 0; ks < BK; ks += 32) {
            v16bf af[4], wf[2], vf[2];
            #pragma unroll
            for (int i = 0; i < 4; ++i)
                af[i] = load_a_frag(&sA[cur][(waveM * 64 + i * 16) * LDK], lane, ks);
            #pragma unroll
            for (int j = 0; j < 2; ++j) {
                wf[j] = load_b_frag(&sW[cur][(waveN * 32 + j * 16) * LDK], lane, ks);
                vf[j] = load_b_frag(&sV[cur][(waveN * 32 + j * 16) * LDK], lane, ks);
            }
            #pragma unroll
            for (int i = 0; i < 4; ++i) {
                #pragma unroll
                for (int j = 0; j < 2; ++j) {
                    accG[i][j] = wmma_bf16(af[i], wf[j], accG[i][j]);
                    accU[i][j] = wmma_bf16(af[i], vf[j], accU[i][j]);
                }
            }
        }
        __syncthreads();   // all waves done reading buf[cur] before it is refilled
    }

    // epilogue: h = silu(gate) * up -> bf16
    const int lm = lane & 15, lh = lane >> 4;
    #pragma unroll
    for (int i = 0; i < 4; ++i) {
        #pragma unroll
        for (int j = 0; j < 2; ++j) {
            const int mBase = mBlock + waveM * 64 + i * 16 + lh * 8;
            const int nCol  = nBlock + waveN * 32 + j * 16 + lm;
            #pragma unroll
            for (int r = 0; r < 8; ++r) {
                const float g = accG[i][j][r];
                const float u = accU[i][j][r];
                const float h = (g / (1.0f + __expf(-g))) * u;
                hbuf[(size_t)(mBase + r) * FFN + nCol] = (__bf16)h;
            }
        }
    }
}

// ---------------------------------------------------------------------------
// Pass 2: out = h @ w2   (contract over FFN), fp32 output
// ---------------------------------------------------------------------------
__global__ __launch_bounds__(256)
void down_kernel(const __bf16* __restrict__ hbuf, const __bf16* __restrict__ w2t,
                 float* __restrict__ out) {
    __shared__ alignas(16) __bf16 sA[2][BM * LDK];
    __shared__ alignas(16) __bf16 sB[2][BN * LDK];

    const int tid   = threadIdx.x;
    const int lane  = tid & 31;
    const int wave  = tid >> 5;
    const int waveM = wave >> 2;
    const int waveN = wave & 3;
    const int mBlock = blockIdx.y * BM;
    const int nBlock = blockIdx.x * BN;

    v8f acc[4][2] = {};

    // 8 async b128 copies per thread per K-step
    auto issue = [&](int buf, int kb) {
        #pragma unroll
        for (int p = 0; p < 4; ++p) {
            const int row = (tid >> 3) + p * 32;
            const int ch  = (tid & 7) * 8;
            async_copy16(&sA[buf][row * LDK + ch],
                         hbuf + (size_t)(mBlock + row) * FFN + kb + ch);
            async_copy16(&sB[buf][row * LDK + ch],
                         w2t  + (size_t)(nBlock + row) * FFN + kb + ch);
        }
    };

    issue(0, 0);
    const int nK = FFN / BK;          // 128
    for (int ki = 0; ki < nK; ++ki) {
        const int cur = ki & 1;
        if (ki + 1 < nK) {
            issue(cur ^ 1, (ki + 1) * BK);
            asm volatile("s_wait_asynccnt 0x8" ::: "memory");
        } else {
            asm volatile("s_wait_asynccnt 0x0" ::: "memory");
        }
        __syncthreads();

        #pragma unroll
        for (int ks = 0; ks < BK; ks += 32) {
            v16bf af[4], bf[2];
            #pragma unroll
            for (int i = 0; i < 4; ++i)
                af[i] = load_a_frag(&sA[cur][(waveM * 64 + i * 16) * LDK], lane, ks);
            #pragma unroll
            for (int j = 0; j < 2; ++j)
                bf[j] = load_b_frag(&sB[cur][(waveN * 32 + j * 16) * LDK], lane, ks);
            #pragma unroll
            for (int i = 0; i < 4; ++i) {
                #pragma unroll
                for (int j = 0; j < 2; ++j)
                    acc[i][j] = wmma_bf16(af[i], bf[j], acc[i][j]);
            }
        }
        __syncthreads();
    }

    const int lm = lane & 15, lh = lane >> 4;
    #pragma unroll
    for (int i = 0; i < 4; ++i) {
        #pragma unroll
        for (int j = 0; j < 2; ++j) {
            const int mBase = mBlock + waveM * 64 + i * 16 + lh * 8;
            const int nCol  = nBlock + waveN * 32 + j * 16 + lm;
            #pragma unroll
            for (int r = 0; r < 8; ++r)
                out[(size_t)(mBase + r) * DMOD + nCol] = acc[i][j][r];
        }
    }
}

extern "C" void kernel_launch(void* const* d_in, const int* in_sizes, int n_in,
                              void* d_out, int out_size, void* d_ws, size_t ws_size,
                              hipStream_t stream) {
    const float* x   = (const float*)d_in[0];
    const float* w1  = (const float*)d_in[1];
    const float* v1  = (const float*)d_in[2];
    const float* w2  = (const float*)d_in[3];
    const int*  eidx = (const int*)d_in[4];
    float* out = (float*)d_out;

    // Workspace layout (bytes): xb 16M | w1b 32M | v1b 32M | w2t 32M | h 64M = 176 MiB
    char* ws = (char*)d_ws;
    __bf16* xb  = (__bf16*)(ws);
    __bf16* w1b = (__bf16*)(ws + (16ull  << 20));
    __bf16* v1b = (__bf16*)(ws + (48ull  << 20));
    __bf16* w2t = (__bf16*)(ws + (80ull  << 20));
    __bf16* hb  = (__bf16*)(ws + (112ull << 20));

    const int nx4 = TSEQ * DMOD / 4;   // 2,097,152
    const int nw4 = FFN * DMOD / 4;    // 4,194,304
    convert_bf16_kernel<<<nx4 / 256, 256, 0, stream>>>(x,  xb,  nx4, eidx, 0);
    convert_bf16_kernel<<<nw4 / 256, 256, 0, stream>>>(w1, w1b, nw4, eidx, (size_t)FFN * DMOD);
    convert_bf16_kernel<<<nw4 / 256, 256, 0, stream>>>(v1, v1b, nw4, eidx, (size_t)FFN * DMOD);
    transpose_convert_kernel<<<dim3(FFN / 32, DMOD / 32), 256, 0, stream>>>(w2, w2t, eidx);

    gateup_kernel<<<dim3(FFN / BN, TSEQ / BM), 256, 0, stream>>>(xb, w1b, v1b, hb);
    down_kernel<<<dim3(DMOD / BN, TSEQ / BM), 256, 0, stream>>>(hb, w2t, out);
}
